// TrueLastDIFMultiHeadAttention_16415365005500
// MI455X (gfx1250) — matrix operand order repaired and checked by reading, not compiled, not run on hardware
//
#include <hip/hip_runtime.h>
#include <hip/hip_bf16.h>

typedef __attribute__((ext_vector_type(16))) _Float16 v16h;
typedef __attribute__((ext_vector_type(8)))  _Float16 v8h;
typedef __attribute__((ext_vector_type(8)))  float    v8f;
typedef __attribute__((ext_vector_type(4)))  float    v4f;
typedef __attribute__((ext_vector_type(4)))  unsigned int u32x4;
typedef __attribute__((ext_vector_type(8)))  int      i32x8;
typedef __attribute__((ext_vector_type(4)))  int      i32x4;

#define B_  32
#define L_  512
#define D_  256
#define H_  4
#define KF_ 160   // fused score K-dim: 64 (q/k) + 64 (qp/kp) + 16 (q0/k0) + 16 (q1/k1)

#define SHUF16(lo, hi) __builtin_shufflevector(lo, hi, 0,1,2,3,4,5,6,7,8,9,10,11,12,13,14,15)

// ---------------------------------------------------------------------------
// One-time packing: W(K x N) fp32 row-major -> WT(N x K) f16 row-major.
// ---------------------------------------------------------------------------
__global__ __launch_bounds__(256)
void pack_wt_kernel(const float* __restrict__ W, _Float16* __restrict__ WT,
                    int kShift, int N)
{
    int idx = blockIdx.x * 256 + threadIdx.x;
    int K = 1 << kShift;
    if (idx >= K * N) return;
    int n = idx >> kShift;
    int k = idx & (K - 1);
    WT[idx] = (_Float16)W[(size_t)k * N + n];
}

// ---------------------------------------------------------------------------
// One-time packing: true_item (B,L,D) fp32 -> Vt (B,H,DH,L) f16 (L-contiguous).
// ---------------------------------------------------------------------------
__global__ __launch_bounds__(256)
void pack_v_kernel(const float* __restrict__ ti, _Float16* __restrict__ Vt)
{
    int idx = blockIdx.x * 256 + threadIdx.x;   // B*H*64*L = 4194304
    int l    = idx & (L_ - 1);
    int rest = idx >> 9;
    int d    = rest & 63;
    int bh   = rest >> 6;
    int b    = bh >> 2;
    int h    = bh & 3;
    Vt[idx] = (_Float16)ti[((size_t)b * L_ + l) * D_ + h * 64 + d];
}

// ---------------------------------------------------------------------------
// WMMA GEMM: C(M x N) = Ain(M x Kin fp32) * WT^T + bias, scattered as f16 into
// fused per-head buffer dst[((b*H+h)*L + l)*KF + dstOff + d].
// ---------------------------------------------------------------------------
__global__ __launch_bounds__(128)
void proj_f16_kernel(const float* __restrict__ Ain, const _Float16* __restrict__ WT,
                     const float* __restrict__ bias, _Float16* __restrict__ dst,
                     int Kin, int phShift, int dstOff, float scale)
{
    const int lane = threadIdx.x & 31;
    const int wave = threadIdx.x >> 5;
    const int l15  = lane & 15;          // A: M row | B/C: N col (within tile)
    const int half = lane >> 4;
    const int m0 = blockIdx.x * 16;
    const int n0 = blockIdx.y * 64 + wave * 16;

    v8f acc = {};
    const float*    arow = Ain + (size_t)(m0 + l15) * Kin;
    const _Float16* wrow = WT  + (size_t)(n0 + l15) * Kin;

    for (int k0 = 0; k0 < Kin; k0 += 32) {
        if (k0 + 32 < Kin) {                       // uniform branch (EXEC stays full)
            __builtin_prefetch(arow + k0 + 32, 0, 3);   // WGP-scope prefetch
            __builtin_prefetch(wrow + k0 + 32, 0, 3);
        }
        // A: two contiguous 8-float chunks -> f16
        v4f f0 = *(const v4f*)(arow + k0 + half * 8);
        v4f f1 = *(const v4f*)(arow + k0 + half * 8 + 4);
        v4f f2 = *(const v4f*)(arow + k0 + 16 + half * 8);
        v4f f3 = *(const v4f*)(arow + k0 + 16 + half * 8 + 4);
        v16h a;
#pragma unroll
        for (int e = 0; e < 4; ++e) {
            a[e]      = (_Float16)f0[e];
            a[4 + e]  = (_Float16)f1[e];
            a[8 + e]  = (_Float16)f2[e];
            a[12 + e] = (_Float16)f3[e];
        }
        // B: two contiguous 8-half chunks (WT is K-contiguous per column)
        v8h b0 = *(const v8h*)(wrow + k0 + half * 8);
        v8h b1 = *(const v8h*)(wrow + k0 + 16 + half * 8);
        v16h bf = SHUF16(b0, b1);
        acc = __builtin_amdgcn_wmma_f32_16x16x32_f16(false, a, false, bf,
                                                     (short)0, acc, false, false);
    }

    const int colN = n0 + l15;
    const float bval = bias[colN];
    const int hh = colN >> phShift;
    const int dd = colN & ((1 << phShift) - 1);
#pragma unroll
    for (int i = 0; i < 8; ++i) {
        int r    = m0 + half * 8 + i;
        int bidx = r >> 9;        // / L
        int lidx = r & (L_ - 1);  // % L
        float v = (acc[i] + bval) * scale;
        dst[(((size_t)bidx * H_ + hh) * L_ + lidx) * KF_ + dstOff + dd] = (_Float16)v;
    }
}

// ---------------------------------------------------------------------------
// Attention: per block = (b, h, 16-query tile).
//  * Q tile (16x160 f16) DMA'd into LDS once per workgroup via the Tensor
//    Data Mover (D# built per ISA 8.3-8.6), waited with s_wait_tensorcnt.
//  * Scores (fused K=160) -> LDS, fp32 softmax with mask,
//  * ctx = P(512) * V(512x64) from packed Vt.
// grid: (L/16, H, B), block: 128
// ---------------------------------------------------------------------------
#define SLDS_STRIDE 516   // 512 + 4 pad

__global__ __launch_bounds__(128)
void attn_kernel(const _Float16* __restrict__ Qf, const _Float16* __restrict__ Kf,
                 const float* __restrict__ mask, const _Float16* __restrict__ Vt,
                 _Float16* __restrict__ ctx)
{
    __shared__ float S[16 * SLDS_STRIDE];               // 33,024 B
    __shared__ __align__(16) _Float16 Qs[16 * KF_];     //  5,120 B

    const int lane = threadIdx.x & 31;
    const int wave = threadIdx.x >> 5;
    const int l15  = lane & 15;
    const int half = lane >> 4;
    const int lt = blockIdx.x;
    const int h  = blockIdx.y;
    const int b  = blockIdx.z;
    const size_t bhL = ((size_t)b * H_ + h) * L_;

    // ---- Stage Q tile into LDS --------------------------------------------
    const _Float16* qbase = Qf + (bhL + (size_t)lt * 16) * KF_;
#if __has_builtin(__builtin_amdgcn_tensor_load_to_lds)
    if (wave == 0) {   // TDM ignores EXEC; issue once per workgroup (uniform)
        unsigned long long ga = (unsigned long long)qbase;       // byte address
        unsigned ldsA = (unsigned)(unsigned long long)&Qs[0];    // addr[31:0] = LDS offset
        // D# group0: count=1 | lds_addr | global_addr[56:0] | type=2
        u32x4 g0 = { 1u, ldsA, (unsigned)ga,
                     (unsigned)((ga >> 32) & 0x01FFFFFFull) | (2u << 30) };
        // D# group1: data_size=2B; tensor_dim0=160, tensor_dim1=512;
        //            tile_dim0=160, tile_dim1=16; tensor_dim0_stride=160
        i32x8 g1 = { (int)0x00010000u,      // workgroup_mask=0, data_size=1(2B)
                     (int)(160u << 16),     // tensor_dim0[15:0]
                     (int)(512u << 16),     // tensor_dim0[31:16]=0 | tensor_dim1[15:0]
                     (int)(160u << 16),     // tensor_dim1[31:16]=0 | tile_dim0=160
                     16,                    // tile_dim1=16, tile_dim2=0
                     160,                   // tensor_dim0_stride[31:0]
                     0, 0 };                // stride hi / tensor_dim1_stride (unused 2D)
        i32x4 gz = { 0, 0, 0, 0 };          // groups 2/3: 2-D tile, unused
#if defined(__clang_major__) && (__clang_major__ >= 23)
        i32x8 gz8 = { 0, 0, 0, 0, 0, 0, 0, 0 };
        __builtin_amdgcn_tensor_load_to_lds(g0, g1, gz, gz, gz8, 0);
#else
        __builtin_amdgcn_tensor_load_to_lds(g0, g1, gz, gz, 0);
#endif
        __builtin_amdgcn_s_wait_tensorcnt(0);
    }
#else
    for (int t = threadIdx.x; t < 16 * KF_; t += 128) Qs[t] = qbase[t];
#endif
    __syncthreads();

    // Preload the 5 K-step A-fragments of Q from LDS (ds_load_b128 pairs).
    v16h qa[5];
    const _Float16* qrow = Qs + l15 * KF_;
#pragma unroll
    for (int kk = 0; kk < 5; ++kk) {
        v8h q0v = *(const v8h*)(qrow + kk * 32 + half * 8);
        v8h q1v = *(const v8h*)(qrow + kk * 32 + 16 + half * 8);
        qa[kk] = SHUF16(q0v, q1v);
    }

    // Scores: wave w handles key-tiles [8w, 8w+8).
    for (int mt = wave * 8; mt < wave * 8 + 8; ++mt) {
        v8f acc = {};
        const _Float16* krow = Kf + (bhL + mt * 16 + l15) * KF_;
        __builtin_prefetch(krow + KF_ * 16, 0, 3);   // next key tile, same lane slot
#pragma unroll
        for (int kk = 0; kk < 5; ++kk) {
            v8h k0v = *(const v8h*)(krow + kk * 32 + half * 8);
            v8h k1v = *(const v8h*)(krow + kk * 32 + 16 + half * 8);
            v16h bf = SHUF16(k0v, k1v);
            acc = __builtin_amdgcn_wmma_f32_16x16x32_f16(false, qa[kk], false, bf,
                                                         (short)0, acc, false, false);
        }
#pragma unroll
        for (int i = 0; i < 8; ++i) {
            int r = half * 8 + i;
            int c = mt * 16 + l15;
            S[r * SLDS_STRIDE + c] = acc[i] + mask[(size_t)(lt * 16 + r) * L_ + c];
        }
    }
    __syncthreads();

    // Softmax: one thread per row (fp32, in LDS).
    if (threadIdx.x < 16) {
        float* row = S + threadIdx.x * SLDS_STRIDE;
        float mx = -3.402823466e+38f;
        for (int c = 0; c < L_; ++c) mx = fmaxf(mx, row[c]);
        float sum = 0.f;
        for (int c = 0; c < L_; ++c) { float e = __expf(row[c] - mx); row[c] = e; sum += e; }
        float inv = 1.f / sum;
        for (int c = 0; c < L_; ++c) row[c] *= inv;
    }
    __syncthreads();

    // ctx: wave w computes d-slice [16w, 16w+16) of DH=64. K = 512.
    const int d0 = wave * 16;
    const _Float16* vrow = Vt + ((size_t)(b * H_ + h) * 64 + d0 + l15) * L_;
    v8f acc = {};
    for (int k0 = 0; k0 < L_; k0 += 32) {
        if (k0 + 32 < L_) __builtin_prefetch(vrow + k0 + 32, 0, 3);
        // A: probs from LDS (fp32 -> f16), two contiguous 8-float chunks
        const float* srow = S + l15 * SLDS_STRIDE + k0;
        v4f s0 = *(const v4f*)(srow + half * 8);
        v4f s1 = *(const v4f*)(srow + half * 8 + 4);
        v4f s2 = *(const v4f*)(srow + 16 + half * 8);
        v4f s3 = *(const v4f*)(srow + 16 + half * 8 + 4);
        v16h pa;
#pragma unroll
        for (int e = 0; e < 4; ++e) {
            pa[e]      = (_Float16)s0[e];
            pa[4 + e]  = (_Float16)s1[e];
            pa[8 + e]  = (_Float16)s2[e];
            pa[12 + e] = (_Float16)s3[e];
        }
        // B: V from packed Vt, contiguous along K (=L)
        v8h vv0 = *(const v8h*)(vrow + k0 + half * 8);
        v8h vv1 = *(const v8h*)(vrow + k0 + 16 + half * 8);
        v16h vb = SHUF16(vv0, vv1);
        acc = __builtin_amdgcn_wmma_f32_16x16x32_f16(false, pa, false, vb,
                                                     (short)0, acc, false, false);
    }
#pragma unroll
    for (int i = 0; i < 8; ++i) {
        int r = half * 8 + i;
        ctx[((size_t)b * L_ + lt * 16 + r) * D_ + h * 64 + d0 + l15] = (_Float16)acc[i];
    }
}

// ---------------------------------------------------------------------------
// Output projection: H(16384x256) = ctx_f16 * Wd + bd (fp32 out).
// ---------------------------------------------------------------------------
__global__ __launch_bounds__(128)
void outproj_kernel(const _Float16* __restrict__ Ain, const _Float16* __restrict__ WT,
                    const float* __restrict__ bias, float* __restrict__ Hout)
{
    const int lane = threadIdx.x & 31;
    const int wave = threadIdx.x >> 5;
    const int l15  = lane & 15;
    const int half = lane >> 4;
    const int m0 = blockIdx.x * 16;
    const int n0 = blockIdx.y * 64 + wave * 16;

    v8f acc = {};
    const _Float16* arow = Ain + (size_t)(m0 + l15) * D_;
    const _Float16* wrow = WT  + (size_t)(n0 + l15) * D_;
    for (int k0 = 0; k0 < D_; k0 += 32) {
        if (k0 + 32 < D_) {
            __builtin_prefetch(arow + k0 + 32, 0, 3);
            __builtin_prefetch(wrow + k0 + 32, 0, 3);
        }
        v8h a0 = *(const v8h*)(arow + k0 + half * 8);
        v8h a1 = *(const v8h*)(arow + k0 + 16 + half * 8);
        v16h a = SHUF16(a0, a1);
        v8h b0 = *(const v8h*)(wrow + k0 + half * 8);
        v8h b1 = *(const v8h*)(wrow + k0 + 16 + half * 8);
        v16h bf = SHUF16(b0, b1);
        acc = __builtin_amdgcn_wmma_f32_16x16x32_f16(false, a, false, bf,
                                                     (short)0, acc, false, false);
    }
    const int c = n0 + l15;
    const float bval = bias[c];
#pragma unroll
    for (int i = 0; i < 8; ++i)
        Hout[(size_t)(m0 + half * 8 + i) * D_ + c] = acc[i] + bval;
}

// ---------------------------------------------------------------------------
// LayerNorm + residual: out = g*(h-mu)/sqrt(var+eps) + b + x, per row of 256.
// ---------------------------------------------------------------------------
__global__ __launch_bounds__(256)
void ln_kernel(const float* __restrict__ Hin, const float* __restrict__ x,
               const float* __restrict__ g, const float* __restrict__ bb,
               float* __restrict__ out)
{
    int row = blockIdx.x * blockDim.x + threadIdx.x;
    if (row >= B_ * L_) return;
    const float* hr = Hin + (size_t)row * D_;
    float s1 = 0.f, s2 = 0.f;
    for (int i = 0; i < D_; i += 4) {
        v4f v = *(const v4f*)(hr + i);
#pragma unroll
        for (int e = 0; e < 4; ++e) { s1 += v[e]; s2 += v[e] * v[e]; }
    }
    float mu  = s1 * (1.0f / D_);
    float var = s2 * (1.0f / D_) - mu * mu;
    float inv = rsqrtf(var + 1e-12f);
    const float* xr = x + (size_t)row * D_;
    float* orow = out + (size_t)row * D_;
    for (int i = 0; i < D_; ++i)
        orow[i] = g[i] * (hr[i] - mu) * inv + bb[i] + xr[i];
}

// ---------------------------------------------------------------------------
extern "C" void kernel_launch(void* const* d_in, const int* in_sizes, int n_in,
                              void* d_out, int out_size, void* d_ws, size_t ws_size,
                              hipStream_t stream)
{
    (void)in_sizes; (void)n_in; (void)out_size; (void)ws_size;

    const float* x         = (const float*)d_in[0];
    const float* pos       = (const float*)d_in[1];
    const float* true_item = (const float*)d_in[2];
    const float* attr0     = (const float*)d_in[3];   // (B,L,1,64) == (16384,64)
    const float* attr1     = (const float*)d_in[4];
    const float* mask      = (const float*)d_in[5];   // (1,1,L,L)
    const float* Wq  = (const float*)d_in[6];   const float* bq  = (const float*)d_in[7];
    const float* Wk  = (const float*)d_in[8];   const float* bk  = (const float*)d_in[9];
    const float* Wqp = (const float*)d_in[10];  const float* bqp = (const float*)d_in[11];
    const float* Wkp = (const float*)d_in[12];  const float* bkp = (const float*)d_in[13];
    const float* Wq0 = (const float*)d_in[14];  const float* bq0 = (const float*)d_in[15];
    const float* Wk0 = (const float*)d_in[16];  const float* bk0 = (const float*)d_in[17];
    const float* Wq1 = (const float*)d_in[18];  const float* bq1 = (const float*)d_in[19];
    const float* Wk1 = (const float*)d_in[20];  const float* bk1 = (const float*)d_in[21];
    const float* Wd  = (const float*)d_in[22];  const float* bd  = (const float*)d_in[23];
    const float* ln_g = (const float*)d_in[24]; const float* ln_b = (const float*)d_in[25];
    float* out = (float*)d_out;

    // Workspace carve-up
    char* ws = (char*)d_ws;
    const size_t qfBytes  = (size_t)B_ * H_ * L_ * KF_ * sizeof(_Float16); // 20 MB
    const size_t ctxBytes = (size_t)B_ * L_ * D_ * sizeof(_Float16);       //  8 MB
    const size_t vtBytes  = (size_t)B_ * H_ * 64 * L_ * sizeof(_Float16);  //  8 MB
    const size_t hBytes   = (size_t)B_ * L_ * D_ * sizeof(float);          // 16 MB
    size_t off = 0;
    _Float16* Qf   = (_Float16*)(ws + off); off += qfBytes;
    _Float16* Kf   = (_Float16*)(ws + off); off += qfBytes;
    _Float16* ctxf = (_Float16*)(ws + off); off += ctxBytes;
    _Float16* Vt   = (_Float16*)(ws + off); off += vtBytes;
    float*    hbuf = (float*)   (ws + off); off += hBytes;
    const size_t wtD = (size_t)D_ * D_;   // 65536 halfs
    const size_t wtA = (size_t)64 * 64;   //  4096 halfs
    _Float16* WqT  = (_Float16*)(ws + off);
    _Float16* WkT  = WqT  + wtD;
    _Float16* WqpT = WkT  + wtD;
    _Float16* WkpT = WqpT + wtD;
    _Float16* WdT  = WkpT + wtD;
    _Float16* Wq0T = WdT  + wtD;
    _Float16* Wk0T = Wq0T + wtA;
    _Float16* Wq1T = Wk0T + wtA;
    _Float16* Wk1T = Wq1T + wtA;

    const dim3 blk(128);
    const dim3 gridD(1024, 4);   // (16384/16, 256/64)
    const dim3 gridA(1024, 1);   // (16384/16,  64/64)
    const float sc = 0.125f;     // 1/sqrt(DH)

    // Pack weights (transposed f16) and V (per-head transposed f16)
    pack_wt_kernel<<<dim3(256), dim3(256), 0, stream>>>(Wq,  WqT,  8, D_);
    pack_wt_kernel<<<dim3(256), dim3(256), 0, stream>>>(Wk,  WkT,  8, D_);
    pack_wt_kernel<<<dim3(256), dim3(256), 0, stream>>>(Wqp, WqpT, 8, D_);
    pack_wt_kernel<<<dim3(256), dim3(256), 0, stream>>>(Wkp, WkpT, 8, D_);
    pack_wt_kernel<<<dim3(256), dim3(256), 0, stream>>>(Wd,  WdT,  8, D_);
    pack_wt_kernel<<<dim3(16),  dim3(256), 0, stream>>>(Wq0, Wq0T, 6, 64);
    pack_wt_kernel<<<dim3(16),  dim3(256), 0, stream>>>(Wk0, Wk0T, 6, 64);
    pack_wt_kernel<<<dim3(16),  dim3(256), 0, stream>>>(Wq1, Wq1T, 6, 64);
    pack_wt_kernel<<<dim3(16),  dim3(256), 0, stream>>>(Wk1, Wk1T, 6, 64);
    pack_v_kernel<<<dim3(16384), dim3(256), 0, stream>>>(true_item, Vt);

    // Fused Q-side (scale folded) and K-side projections
    proj_f16_kernel<<<gridD, blk, 0, stream>>>(x,     WqT,  bq,  Qf, 256, 6,   0, sc);
    proj_f16_kernel<<<gridD, blk, 0, stream>>>(pos,   WqpT, bqp, Qf, 256, 6,  64, sc);
    proj_f16_kernel<<<gridA, blk, 0, stream>>>(attr0, Wq0T, bq0, Qf,  64, 4, 128, sc);
    proj_f16_kernel<<<gridA, blk, 0, stream>>>(attr1, Wq1T, bq1, Qf,  64, 4, 144, sc);
    proj_f16_kernel<<<gridD, blk, 0, stream>>>(x,     WkT,  bk,  Kf, 256, 6,   0, 1.f);
    proj_f16_kernel<<<gridD, blk, 0, stream>>>(pos,   WkpT, bkp, Kf, 256, 6,  64, 1.f);
    proj_f16_kernel<<<gridA, blk, 0, stream>>>(attr0, Wk0T, bk0, Kf,  64, 4, 128, 1.f);
    proj_f16_kernel<<<gridA, blk, 0, stream>>>(attr1, Wk1T, bk1, Kf,  64, 4, 144, 1.f);

    // Attention (TDM Q-stage + scores + softmax + ctx)
    attn_kernel<<<dim3(L_ / 16, H_, B_), blk, 0, stream>>>(Qf, Kf, mask, Vt, ctxf);

    // Output projection, then LayerNorm + residual
    outproj_kernel<<<gridD, blk, 0, stream>>>(ctxf, WdT, bd, hbuf);
    ln_kernel<<<dim3((B_ * L_ + 255) / 256), dim3(256), 0, stream>>>(hbuf, x, ln_g, ln_b, out);
}
